// Attn2d_35381940584836
// MI455X (gfx1250) — matrix operand order repaired and checked by reading, not compiled, not run on hardware
//
#include <hip/hip_runtime.h>

// ---------------- problem constants ----------------
#define CIN    256
#define F_OUT  256
#define NH     8
#define DH     32
#define KK     64      // 8x8 tile positions
#define HQ     16
#define WQ     16
#define BATCH  8
#define SCALE  0.42044820762685725f   // 32^(-0.25)

typedef __bf16 bf16_t;
typedef __attribute__((ext_vector_type(16))) __bf16 v16bf;
typedef __attribute__((ext_vector_type(8)))  float  v8f;

// ---------------- LDS layout (no static LDS -> dynamic base offset is 0) ----
#define WSTR   264                     // padded stride (bf16) for 256-wide K rows
#define KVSTR  68                      // padded stride (f32) for key/val rows
#define OFF_W    0
#define SZ_W     (256 * WSTR * 2)      // 135168 B : weight tile (bf16)
#define OFF_X    (OFF_W + SZ_W)        // 135168
#define SZ_X     (64 * WSTR * 2)       //  33792 B : x tile [pos][ch] (bf16)
#define OFF_KV   (OFF_X + SZ_X)        // 168960
#define SZ_KV    (256 * KVSTR * 4)     //  69632 B : key/val [ch][pos] (f32)
#define OFF_Q    (OFF_KV + SZ_KV)      // 238592
#define SZ_Q     (256 * 4)
#define OFF_LG   (OFF_Q + SZ_Q)        // 239616
#define SZ_LG    (NH * KK * 4)         //   2048 B
#define OFF_POS  (OFF_LG + SZ_LG)      // 241664
#define SZ_POS   (256 * 64 * 4)        //  65536 B : pos_feats staged in LDS
#define OFF_PB   (OFF_POS + SZ_POS)    // 307200
#define SZ_PB    (768 * 4)             //   3072 B : proj_b staged in LDS
#define SMEM_BYTES (OFF_PB + SZ_PB)    // 310272 B (fits 320KB WGP LDS)

// ---------------- CDNA5 async global->LDS (ASYNCcnt-tracked) ----------------
// Copies 16B per active lane: LDS[lds_off] = MEM[gbase + goff]
__device__ __forceinline__ void async_ld16(uint32_t lds_off, uint64_t gbase,
                                           uint32_t goff) {
  asm volatile("global_load_async_to_lds_b128 %0, %1, %2"
               :: "v"(lds_off), "v"(goff), "s"(gbase)
               : "memory");
}
__device__ __forceinline__ void wait_async0() {
  asm volatile("s_wait_asynccnt 0x0" ::: "memory");
}

// Stage 256x256 bf16 weight block (rows contiguous, 512B) into padded LDS tile.
__device__ __forceinline__ void stage_weights_async(const bf16_t* __restrict__ wsrc,
                                                    int tid) {
  const uint64_t gbase = (uint64_t)(uintptr_t)wsrc;
  #pragma unroll 4
  for (int g = tid; g < 8192; g += 256) {       // 8192 x 16B chunks
    const int row = g >> 5;
    const int ch  = g & 31;
    const uint32_t lds_off = OFF_W + (uint32_t)row * (WSTR * 2) + ch * 16;
    const uint32_t goff    = (uint32_t)(row * 512 + ch * 16);
    async_ld16(lds_off, gbase, goff);
  }
}

// ---------------- WMMA helpers ----------------
__device__ __forceinline__ v8f wmma_bf16(v16bf a, v16bf b, v8f c) {
  return __builtin_amdgcn_wmma_f32_16x16x32_bf16(
      /*neg_a=*/false, a, /*neg_b=*/false, b,
      /*c_mod=*/(short)0, c, /*reuse_a=*/false, /*reuse_b=*/false);
}

// A-fragment (16x32 bf16): lane half interleaves K blocks of 8
__device__ __forceinline__ v16bf load_fragA(const bf16_t* base, int row, int kb) {
  const bf16_t* p = base + row * WSTR + kb;
  union { uint4 q[2]; v16bf v; } u;
  u.q[0] = *reinterpret_cast<const uint4*>(p);
  u.q[1] = *reinterpret_cast<const uint4*>(p + 16);
  return u.v;
}
// B-fragment (32x16 bf16): lane half interleaves K blocks of 8
__device__ __forceinline__ v16bf load_fragB(const bf16_t* base, int row, int kb) {
  const bf16_t* p = base + row * WSTR + kb;
  union { uint4 q[2]; v16bf v; } u;
  u.q[0] = *reinterpret_cast<const uint4*>(p);
  u.q[1] = *reinterpret_cast<const uint4*>(p + 16);
  return u.v;
}

// 256x64 = (256x256)x(256x64): wave owns 32 M-rows (2 tiles) x 4 N-tiles
__device__ __forceinline__ void gemm_tile(const bf16_t* __restrict__ wlds,
                                          const bf16_t* __restrict__ xt,
                                          int lane, int wave, v8f (&acc)[2][4]) {
  const int l15 = lane & 15;
  const int hi  = lane >> 4;
  const int m0  = wave * 32;
  #pragma unroll
  for (int kk = 0; kk < CIN; kk += 32) {
    const int kbA = kk + (hi << 3);
    const int kbB = kk + (hi << 3);
    v16bf a0 = load_fragA(wlds, m0 + l15,      kbA);
    v16bf a1 = load_fragA(wlds, m0 + 16 + l15, kbA);
    v16bf b0 = load_fragB(xt, l15,      kbB);
    v16bf b1 = load_fragB(xt, 16 + l15, kbB);
    v16bf b2 = load_fragB(xt, 32 + l15, kbB);
    v16bf b3 = load_fragB(xt, 48 + l15, kbB);
    acc[0][0] = wmma_bf16(a0, b0, acc[0][0]);
    acc[0][1] = wmma_bf16(a0, b1, acc[0][1]);
    acc[0][2] = wmma_bf16(a0, b2, acc[0][2]);
    acc[0][3] = wmma_bf16(a0, b3, acc[0][3]);
    acc[1][0] = wmma_bf16(a1, b0, acc[1][0]);
    acc[1][1] = wmma_bf16(a1, b1, acc[1][1]);
    acc[1][2] = wmma_bf16(a1, b2, acc[1][2]);
    acc[1][3] = wmma_bf16(a1, b3, acc[1][3]);
  }
}

// ---------------- pre-pass: proj_w fp32 -> bf16 in workspace ----------------
__global__ void convert_w_kernel(const float* __restrict__ w,
                                 bf16_t* __restrict__ o, int n) {
  for (int i = blockIdx.x * blockDim.x + threadIdx.x; i < n;
       i += gridDim.x * blockDim.x)
    o[i] = (bf16_t)w[i];
}

// ---------------- fused qkv-proj + tile attention ----------------
__global__ __launch_bounds__(256)
void attn2d_fused_kernel(const float* __restrict__ x,
                         const bf16_t* __restrict__ wbf,   // [768][256] bf16
                         const float* __restrict__ pb,     // [768]
                         const float* __restrict__ pos,    // [256][64]
                         float* __restrict__ out) {        // [B][256][16][16]
  extern __shared__ char smem[];
  bf16_t* wlds  = (bf16_t*)(smem + OFF_W);
  bf16_t* xt    = (bf16_t*)(smem + OFF_X);
  float*  kv    = (float*)(smem + OFF_KV);
  float*  qs    = (float*)(smem + OFF_Q);
  float*  lg    = (float*)(smem + OFF_LG);
  float*  plds  = (float*)(smem + OFF_POS);
  float*  pblds = (float*)(smem + OFF_PB);

  const int tid  = threadIdx.x;
  const int lane = tid & 31;          // wave32
  const int wave = tid >> 5;          // 8 waves
  const int wq = blockIdx.x, hq = blockIdx.y, b = blockIdx.z;

  // ---- async stage: K-weights, pos_feats, proj_b straight into LDS ----
  stage_weights_async(wbf + 256 * CIN, tid);
  {
    const uint64_t gpos = (uint64_t)(uintptr_t)pos;
    #pragma unroll 4
    for (int g = tid; g < 4096; g += 256)       // 64KB = 4096 x 16B
      async_ld16(OFF_POS + (uint32_t)g * 16, gpos, (uint32_t)g * 16);
    const uint64_t gpb = (uint64_t)(uintptr_t)pb;
    if (tid < 192)                              // 3KB = 192 x 16B
      async_ld16(OFF_PB + (uint32_t)tid * 16, gpb, (uint32_t)tid * 16);
  }

  // ---- stage x tile (needs fp32->bf16 cvt): [64 pos][256 ch] ----
  {
    const int c = tid;
    const float* xbase =
        x + (((size_t)b * CIN + c) * 128 + hq * 8) * 128 + (size_t)wq * 8;
    #pragma unroll
    for (int i = 0; i < 8; ++i) {
      const float* xr = xbase + (size_t)i * 128;
      float4 r0 = *reinterpret_cast<const float4*>(xr);
      float4 r1 = *reinterpret_cast<const float4*>(xr + 4);
      float v[8] = {r0.x, r0.y, r0.z, r0.w, r1.x, r1.y, r1.z, r1.w};
      #pragma unroll
      for (int j = 0; j < 8; ++j)
        xt[(i * 8 + j) * WSTR + c] = (bf16_t)v[j];
    }
  }
  // warm L2 with the V-projection weights for phase B
  __builtin_prefetch(wbf + 512 * CIN + (size_t)tid * 512, 0, 1);
  wait_async0();
  __syncthreads();

  // ---- q at tile origin (pos 0): per-thread 256-MAC dot, bf16 inputs ----
  {
    float acc = pblds[tid];
    const bf16_t* wrow = wbf + (size_t)tid * CIN;
    #pragma unroll 8
    for (int c = 0; c < CIN; ++c)
      acc += (float)wrow[c] * (float)xt[c];
    qs[tid] = SCALE * acc;
  }

  const v8f vzero = {0.f, 0.f, 0.f, 0.f, 0.f, 0.f, 0.f, 0.f};

  // ---- phase A: key GEMM + (pos_feats + scale*(k+bias)) epilogue ----
  {
    v8f acc[2][4];
    #pragma unroll
    for (int i = 0; i < 2; ++i)
      #pragma unroll
      for (int j = 0; j < 4; ++j) acc[i][j] = vzero;
    gemm_tile(wlds, xt, lane, wave, acc);

    const int l15 = lane & 15, hi = lane >> 4;
    #pragma unroll
    for (int it = 0; it < 2; ++it) {
      const int mb = wave * 32 + it * 16 + hi * 8;     // C layout: VGPR i -> M=mb+i
      #pragma unroll
      for (int nt = 0; nt < 4; ++nt) {
        union { v8f v; float f[8]; } u; u.v = acc[it][nt];
        const int n = nt * 16 + l15;                    // N = lane&15
        #pragma unroll
        for (int i = 0; i < 8; ++i) {
          const int m = mb + i;
          kv[m * KVSTR + n] = plds[m * 64 + n] + SCALE * (u.f[i] + pblds[256 + m]);
        }
      }
    }
  }
  __syncthreads();

  // ---- async restage V weights (overlaps logits+softmax), then logits ----
  stage_weights_async(wbf + 512 * CIN, tid);
  {
    #pragma unroll
    for (int pair = tid; pair < NH * KK; pair += 256) {
      const int h = pair >> 6, k = pair & 63;
      float a = 0.f;
      #pragma unroll 8
      for (int d = 0; d < DH; ++d)
        a += qs[h * DH + d] * kv[(h * DH + d) * KVSTR + k];
      lg[pair] = a;
    }
  }
  __syncthreads();

  // ---- softmax over KK=64: wave w owns head w, wave32 shuffles ----
  {
    const int h = wave;
    float x0 = lg[h * KK + lane];
    float x1 = lg[h * KK + 32 + lane];
    float mx = fmaxf(x0, x1);
    #pragma unroll
    for (int off = 16; off > 0; off >>= 1)
      mx = fmaxf(mx, __shfl_xor(mx, off, 32));
    float e0 = __expf(x0 - mx), e1 = __expf(x1 - mx);
    float s = e0 + e1;
    #pragma unroll
    for (int off = 16; off > 0; off >>= 1)
      s += __shfl_xor(s, off, 32);
    const float inv = 1.0f / s;
    lg[h * KK + lane]      = e0 * inv;
    lg[h * KK + 32 + lane] = e1 * inv;
  }
  wait_async0();            // this wave's V-weight copies are in LDS
  __syncthreads();          // ...and every other wave's too

  // ---- phase B: value GEMM (bias only) ----
  {
    v8f acc[2][4];
    #pragma unroll
    for (int i = 0; i < 2; ++i)
      #pragma unroll
      for (int j = 0; j < 4; ++j) acc[i][j] = vzero;
    gemm_tile(wlds, xt, lane, wave, acc);

    const int l15 = lane & 15, hi = lane >> 4;
    #pragma unroll
    for (int it = 0; it < 2; ++it) {
      const int mb = wave * 32 + it * 16 + hi * 8;
      #pragma unroll
      for (int nt = 0; nt < 4; ++nt) {
        union { v8f v; float f[8]; } u; u.v = acc[it][nt];
        const int n = nt * 16 + l15;
        #pragma unroll
        for (int i = 0; i < 8; ++i) {
          const int m = mb + i;
          kv[m * KVSTR + n] = u.f[i] + pblds[512 + m];
        }
      }
    }
  }
  __syncthreads();

  // ---- out[b, o=tid, hq, wq] = sum_k attn[h][k] * val[o][k] ----
  {
    const int h = tid >> 5;   // tid = h*32 + d
    float a = 0.f;
    #pragma unroll 8
    for (int k = 0; k < KK; ++k)
      a += lg[h * KK + k] * kv[tid * KVSTR + k];
    out[(((size_t)b * F_OUT + tid) * HQ + hq) * WQ + wq] = a;
  }
}

// ---------------- launcher ----------------
extern "C" void kernel_launch(void* const* d_in, const int* in_sizes, int n_in,
                              void* d_out, int out_size, void* d_ws, size_t ws_size,
                              hipStream_t stream) {
  (void)in_sizes; (void)n_in; (void)out_size; (void)ws_size;
  const float* x   = (const float*)d_in[0];
  const float* pw  = (const float*)d_in[1];  // (768,256)
  const float* pb  = (const float*)d_in[2];  // (768,)
  const float* pos = (const float*)d_in[3];  // (256,64)
  float* out = (float*)d_out;
  bf16_t* wbf = (bf16_t*)d_ws;               // 768*256*2 = 384 KB scratch

  convert_w_kernel<<<dim3(192), dim3(256), 0, stream>>>(pw, wbf, 768 * CIN);

  dim3 grid(WQ, HQ, BATCH);                  // 2048 workgroups
  attn2d_fused_kernel<<<grid, dim3(256), SMEM_BYTES, stream>>>(x, wbf, pb, pos, out);
}
